// MCARMobilenetv2_11991548690571
// MI455X (gfx1250) — matrix operand order repaired and checked by reading, not compiled, not run on hardware
//
#include <hip/hip_runtime.h>
#include <hip/hip_bf16.h>
#include <math.h>

// ---------------- problem constants ----------------
#define BB   8
#define HH   448
#define WW   448
#define NC   80
#define NF   1280
#define TOPN 4
#define OH   14
#define OW   14
#define PIX  196            // 14*14
#define KDIM 27             // 3*3*3 im2col depth
#define KPAD 32             // padded to one WMMA K-step
#define NGLOB (BB*PIX)      // 1568 = 98 * 16
#define NLOC  (BB*TOPN*PIX) // 6272 = 392 * 16

typedef __attribute__((ext_vector_type(8)))  float    v8f;
typedef __attribute__((ext_vector_type(16))) _Float16 v16h;

__device__ __forceinline__ float sigmoidf_(float x) { return 1.0f / (1.0f + expf(-x)); }

// im2col element for the global (un-cropped) conv
__device__ __forceinline__ float im2col_g(const float* __restrict__ in,
                                          int k, int b, int oy, int ox) {
  if (k >= KDIM) return 0.0f;
  int c  = k / 9, r = k % 9, ky = r / 3, kx = r % 3;
  int iy = oy * 32 - 1 + ky;
  int ix = ox * 32 - 1 + kx;
  if ((unsigned)iy >= (unsigned)HH || (unsigned)ix >= (unsigned)WW) return 0.0f;
  return in[(((size_t)b * 3 + c) * HH + iy) * WW + ix];
}

// im2col element for the local conv, fused with bilinear crop resampling
__device__ __forceinline__ float im2col_l(const float* __restrict__ in,
                                          int k, int b, int oy, int ox,
                                          float y1f, float ystep, float x1f, float xstep) {
  if (k >= KDIM) return 0.0f;
  int c  = k / 9, r = k % 9, ky = r / 3, kx = r % 3;
  int iy = oy * 32 - 1 + ky;
  int ix = ox * 32 - 1 + kx;
  if ((unsigned)iy >= (unsigned)HH || (unsigned)ix >= (unsigned)WW) return 0.0f;
  float ys  = y1f + (float)iy * ystep;
  float xs  = x1f + (float)ix * xstep;
  float y0f = fminf(fmaxf(floorf(ys), 0.0f), 447.0f);
  float x0f = fminf(fmaxf(floorf(xs), 0.0f), 447.0f);
  int y0 = (int)y0f, x0 = (int)x0f;
  int y1i = min(y0 + 1, 447), x1i = min(x0 + 1, 447);
  float wy = ys - y0f, wx = xs - x0f;
  const float* ch = in + ((size_t)b * 3 + c) * (size_t)HH * WW;
  float a  = ch[(size_t)y0  * WW + x0 ];
  float bv = ch[(size_t)y0  * WW + x1i];
  float cv = ch[(size_t)y1i * WW + x0 ];
  float dv = ch[(size_t)y1i * WW + x1i];
  float top = a  * (1.0f - wx) + bv * wx;
  float bot = cv * (1.0f - wx) + dv * wx;
  return top * (1.0f - wy) + bot * wy;
}

// -------- fp32 -> fp16 converters (weights, converted once, L2 resident) ----
__global__ void k_cvt_f16(const float* __restrict__ src, _Float16* __restrict__ dst, int n) {
  int i = blockIdx.x * blockDim.x + threadIdx.x;
  if (i < n) dst[i] = (_Float16)src[i];
}

// feat_w (NF x 27) -> zero-padded (NF x 32) f16: removes k<27 guards in hot loop
__global__ void k_cvt_pad(const float* __restrict__ src, _Float16* __restrict__ dst, int rows) {
  int i = blockIdx.x * blockDim.x + threadIdx.x;
  if (i >= rows * KPAD) return;
  int m = i >> 5, k = i & 31;
  dst[i] = (k < KDIM) ? (_Float16)src[m * KDIM + k] : (_Float16)0.0f;
}

// ---------------- global conv -----------------------------------------------
// One block (8 waves) per 16-pixel N-tile. B tile (32x16 f16) built once in
// LDS, B fragment register-resident; each wave loops 10 M-tiles (80 total),
// one V_WMMA_F32_16X16X32_F16 per tile. ReLU6 -> ga.
__global__ void __launch_bounds__(256)
k_conv_global(const float* __restrict__ in, const _Float16* __restrict__ fw16p,
              const float* __restrict__ fb, float* __restrict__ ga) {
  __shared__ _Float16 lb[16 * KPAD];               // [n_local][k_local]
  int nt = blockIdx.x;
  int wv = threadIdx.x >> 5;
  int lane = threadIdx.x & 31;
  int hi = lane >> 4, nl = lane & 15;

  // cooperative B fill: 512 samples, once per N-tile (was 80x per tile before)
  for (int idx = threadIdx.x; idx < 16 * KPAD; idx += 256) {
    int nloc = idx >> 5, kloc = idx & 31;
    int nn = nt * 16 + nloc;
    int fb_ = nn / PIX, fp = nn % PIX;
    lb[idx] = (_Float16)im2col_g(in, kloc, fb_, fp / OW, fp % OW);
  }
  __syncthreads();

  v16h bf;                                          // B fragment: reused for all M-tiles
#pragma unroll
  for (int j = 0; j < 8; ++j) {
    int kb = ((j & 4) ? 16 : 0) + hi * 8 + (j & 3) * 2; // lanes 0-15: K 0..7,16..23; 16-31: K 8..15,24..31
    bf[2 * j]     = lb[nl * KPAD + kb];
    bf[2 * j + 1] = lb[nl * KPAD + kb + 1];
  }
  int nn = nt * 16 + nl;
  int b = nn / PIX, p = nn % PIX;

  for (int it = 0; it < 10; ++it) {
    int mt = it * 8 + wv;
    int m = mt * 16 + nl;
    v16h a;
#pragma unroll
    for (int j = 0; j < 8; ++j) {
      int k0 = ((j & 4) ? 16 : 0) + hi * 8 + (j & 3) * 2;
      a[2 * j]     = fw16p[m * KPAD + k0];
      a[2 * j + 1] = fw16p[m * KPAD + k0 + 1];
    }
    v8f acc = {};
    acc = __builtin_amdgcn_wmma_f32_16x16x32_f16(false, a, false, bf, (short)0, acc, false, false);
#pragma unroll
    for (int j = 0; j < 8; ++j) {
      int ch = mt * 16 + hi * 8 + j;                // C/D layout: M = j + 8*hi
      float v = fminf(fmaxf(acc[j] + fb[ch], 0.0f), 6.0f);
      ga[((size_t)b * NF + ch) * PIX + p] = v;
    }
  }
}

// ---------------- local conv (bilinear crop fused into im2col) --------------
__global__ void __launch_bounds__(256)
k_conv_local(const float* __restrict__ in, const _Float16* __restrict__ fw16p,
             const float* __restrict__ fb, const int* __restrict__ boxes,
             float* __restrict__ la) {
  __shared__ _Float16 lb[16 * KPAD];
  int nt = blockIdx.x;
  int wv = threadIdx.x >> 5;
  int lane = threadIdx.x & 31;
  int hi = lane >> 4, nl = lane & 15;

  // cooperative B fill: bilinear crop sampling done once per tile (was 80x)
  for (int idx = threadIdx.x; idx < 16 * KPAD; idx += 256) {
    int nloc = idx >> 5, kloc = idx & 31;
    int nn = nt * 16 + nloc;
    int img = nn / PIX, fp = nn % PIX;
    const int* bx = boxes + img * 4;                // y1,y2,x1,x2
    float y1f = (float)bx[0], x1f = (float)bx[2];
    float ystep = ((float)bx[1] - 1.0f - y1f) * (1.0f / 447.0f);
    float xstep = ((float)bx[3] - 1.0f - x1f) * (1.0f / 447.0f);
    lb[idx] = (_Float16)im2col_l(in, kloc, img / TOPN, fp / OW, fp % OW,
                                 y1f, ystep, x1f, xstep);
  }
  __syncthreads();

  v16h bf;
#pragma unroll
  for (int j = 0; j < 8; ++j) {
    int kb = ((j & 4) ? 16 : 0) + hi * 8 + (j & 3) * 2;
    bf[2 * j]     = lb[nl * KPAD + kb];
    bf[2 * j + 1] = lb[nl * KPAD + kb + 1];
  }
  int nn = nt * 16 + nl;
  int img = nn / PIX, p = nn % PIX;

  for (int it = 0; it < 10; ++it) {
    int mt = it * 8 + wv;
    int m = mt * 16 + nl;
    v16h a;
#pragma unroll
    for (int j = 0; j < 8; ++j) {
      int k0 = ((j & 4) ? 16 : 0) + hi * 8 + (j & 3) * 2;
      a[2 * j]     = fw16p[m * KPAD + k0];
      a[2 * j + 1] = fw16p[m * KPAD + k0 + 1];
    }
    v8f acc = {};
    acc = __builtin_amdgcn_wmma_f32_16x16x32_f16(false, a, false, bf, (short)0, acc, false, false);
#pragma unroll
    for (int j = 0; j < 8; ++j) {
      int ch = mt * 16 + hi * 8 + j;
      float v = fminf(fmaxf(acc[j] + fb[ch], 0.0f), 6.0f);
      la[((size_t)img * NF + ch) * PIX + p] = v;
    }
  }
}

// ---------------- mean over the 196 spatial positions -----------------------
__global__ void k_reduce_mean(const float* __restrict__ src, float* __restrict__ dst, int nrows) {
  int i = blockIdx.x * blockDim.x + threadIdx.x;
  if (i >= nrows) return;
  const float* r = src + (size_t)i * PIX;
  float s = 0.0f;
  for (int p = 0; p < PIX; ++p) s += r[p];
  dst[i] = s * (1.0f / (float)PIX);
}

// ---------------- gs head: sigmoid(gf . cls_w + cls_b) -> d_out[0:640] ------
__global__ void k_head_gs(const float* __restrict__ gf, const float* __restrict__ cw,
                          const float* __restrict__ cb, float* __restrict__ out) {
  int i = blockIdx.x * blockDim.x + threadIdx.x;
  if (i >= BB * NC) return;
  int b = i / NC, c = i % NC;
  const float* f = gf + (size_t)b * NF;
  const float* w = cw + (size_t)c * NF;
  float s = cb[c];
  for (int k = 0; k < NF; ++k) s += f[k] * w[k];
  out[i] = sigmoidf_(s);
}

// ---------------- CAM GEMM: M=80, K=1280, N=1568, WMMA f16, LDS-staged B ----
__global__ void __launch_bounds__(160)
k_cam(const float* __restrict__ ga, const _Float16* __restrict__ cw16,
      const float* __restrict__ cb, float* __restrict__ cam) {
  __shared__ _Float16 lb[16 * 32];                 // B tile: [n_local][k_local]
  int nt = blockIdx.x;
  int wv = threadIdx.x >> 5;                       // 0..4 == M tile
  int lane = threadIdx.x & 31;
  int hi = lane >> 4, nl = lane & 15;
  int mcls = wv * 16 + nl;                         // A row (class)

  v8f acc = {};
  for (int kk = 0; kk < NF; kk += 32) {
    __syncthreads();
    for (int idx = threadIdx.x; idx < 512; idx += 160) {
      int nloc = idx >> 5, kloc = idx & 31;
      int nn = nt * 16 + nloc;
      int b = nn / PIX, p = nn % PIX;
      lb[idx] = (_Float16)ga[((size_t)b * NF + (kk + kloc)) * PIX + p];
    }
    __syncthreads();
    v16h a, bf;
#pragma unroll
    for (int j = 0; j < 8; ++j) {
      int k0 = kk + ((j & 4) ? 16 : 0) + hi * 8 + (j & 3) * 2;
      a[2 * j]     = cw16[(size_t)mcls * NF + k0];
      a[2 * j + 1] = cw16[(size_t)mcls * NF + k0 + 1];
      int kb = ((j & 4) ? 16 : 0) + hi * 8 + (j & 3) * 2;
      bf[2 * j]     = lb[nl * 32 + kb];
      bf[2 * j + 1] = lb[nl * 32 + kb + 1];
    }
    acc = __builtin_amdgcn_wmma_f32_16x16x32_f16(false, a, false, bf, (short)0, acc, false, false);
  }

  int nn = nt * 16 + nl;
  int b = nn / PIX, p = nn % PIX;
#pragma unroll
  for (int j = 0; j < 8; ++j) {
    int cls = wv * 16 + hi * 8 + j;
    cam[((size_t)b * NC + cls) * PIX + p] = sigmoidf_(acc[j] + cb[cls]);
  }
}

// ---------------- fused upsample + row/col max (never materialize 514 MB) ---
__global__ void __launch_bounds__(448)
k_scores(const float* __restrict__ cam, float* __restrict__ wsc, float* __restrict__ hsc) {
  __shared__ float t[PIX];
  int bc = blockIdx.x;                              // 0..639 == b*80+c
  const float* src = cam + (size_t)bc * PIX;
  for (int i = threadIdx.x; i < PIX; i += blockDim.x) t[i] = src[i];
  __syncthreads();

  const float S = 13.0f / 447.0f;
  int x = threadIdx.x;
  float xs  = (float)x * S;
  float x0f = fminf(floorf(xs), 13.0f);
  int   x0  = (int)x0f, x1i = min(x0 + 1, 13);
  float wx  = xs - x0f;

  // wscore: max over all sampled y for this x
  float mx = -1e30f;
  for (int y = 0; y < HH; ++y) {
    float ys  = (float)y * S;
    float y0f = fminf(floorf(ys), 13.0f);
    int   y0  = (int)y0f, y1i = min(y0 + 1, 13);
    float wy  = ys - y0f;
    float top = t[y0  * OW + x0] * (1.0f - wx) + t[y0  * OW + x1i] * wx;
    float bot = t[y1i * OW + x0] * (1.0f - wx) + t[y1i * OW + x1i] * wx;
    mx = fmaxf(mx, top * (1.0f - wy) + bot * wy);
  }
  wsc[(size_t)bc * WW + x] = mx;

  // hscore: this thread's index is y; max over all sampled x
  int y0 = x0, y1i = x1i; float wy = wx;
  mx = -1e30f;
  for (int xx = 0; xx < WW; ++xx) {
    float xv  = (float)xx * S;
    float xf  = fminf(floorf(xv), 13.0f);
    int   a0  = (int)xf, a1 = min(a0 + 1, 13);
    float wxx = xv - xf;
    float top = t[y0  * OW + a0] * (1.0f - wxx) + t[y0  * OW + a1] * wxx;
    float bot = t[y1i * OW + a0] * (1.0f - wxx) + t[y1i * OW + a1] * wxx;
    mx = fmaxf(mx, top * (1.0f - wy) + bot * wy);
  }
  hsc[(size_t)bc * HH + x] = mx;
}

// ---------------- top-4 select + minmax_norm + obj_loc -> 32 boxes ----------
__global__ void k_boxes(const float* __restrict__ gs, const float* __restrict__ wsc,
                        const float* __restrict__ hsc, int* __restrict__ boxes) {
  int tid = threadIdx.x;
  if (tid >= BB * TOPN) return;
  int b = tid / TOPN, t = tid % TOPN;

  // rank-select t-th largest (ties -> lower index first, matching top_k)
  const float* g = gs + (size_t)b * NC;
  int cls = 0;
  for (int c = 0; c < NC; ++c) {
    int rank = 0;
    float v = g[c];
    for (int c2 = 0; c2 < NC; ++c2) {
      float v2 = g[c2];
      if (v2 > v || (v2 == v && c2 < c)) rank++;
    }
    if (rank == t) { cls = c; break; }
  }

  const float* sy = hsc + (size_t)(b * NC + cls) * HH;
  const float* sx = wsc + (size_t)(b * NC + cls) * WW;
  int res[4];
  for (int axis = 0; axis < 2; ++axis) {
    const float* s = (axis == 0) ? sy : sx;
    float mn = 1e30f, mxv = -1e30f;
    for (int i = 0; i < 448; ++i) { float v = s[i]; mn = fminf(mn, v); mxv = fmaxf(mxv, v); }
    float rng = mxv - mn;
    bool useRng = (rng > 0.0f);
    float denom = useRng ? rng : ((mxv != 0.0f) ? mxv : 1.0f);
    int first = -1, last = -1;
    for (int i = 0; i < 448; ++i) {
      float nv = useRng ? (s[i] - mn) / denom : s[i] / denom;
      if (nv >= 0.5f) { if (first < 0) first = i; last = i; }
    }
    int lo, hi;
    if (first >= 0) { lo = first; hi = last + 1; }
    else            { lo = 56;    hi = 392;      }   // 448*0.125, 448*0.875
    int need = 56 - (hi - lo); if (need < 0) need = 0;
    lo -= (need + 1) / 2; if (lo < 0) lo = 0;
    hi += (need + 1) / 2; if (hi > 448) hi = 448;
    res[axis * 2 + 0] = lo; res[axis * 2 + 1] = hi;
  }
  boxes[tid * 4 + 0] = res[0];  // y1
  boxes[tid * 4 + 1] = res[1];  // y2
  boxes[tid * 4 + 2] = res[2];  // x1
  boxes[tid * 4 + 3] = res[3];  // x2
}

// ---------------- ls head: max over 4 crops -> d_out[640:1280] --------------
__global__ void k_head_ls(const float* __restrict__ lf, const float* __restrict__ cw,
                          const float* __restrict__ cb, float* __restrict__ out) {
  int i = blockIdx.x * blockDim.x + threadIdx.x;
  if (i >= BB * NC) return;
  int b = i / NC, c = i % NC;
  const float* w = cw + (size_t)c * NF;
  float best = -1e30f;
  for (int t = 0; t < TOPN; ++t) {
    const float* f = lf + (size_t)(b * TOPN + t) * NF;
    float s = cb[c];
    for (int k = 0; k < NF; ++k) s += f[k] * w[k];
    best = fmaxf(best, sigmoidf_(s));
  }
  out[BB * NC + i] = best;
}

// ---------------- launch --------------------------------------------------
static inline size_t align256(size_t x) { return (x + 255) & ~(size_t)255; }

extern "C" void kernel_launch(void* const* d_in, const int* in_sizes, int n_in,
                              void* d_out, int out_size, void* d_ws, size_t ws_size,
                              hipStream_t stream) {
  (void)in_sizes; (void)n_in; (void)out_size; (void)ws_size;
  const float* in = (const float*)d_in[0];
  const float* fw = (const float*)d_in[1];
  const float* fb = (const float*)d_in[2];
  const float* cw = (const float*)d_in[3];
  const float* cb = (const float*)d_in[4];
  float* out = (float*)d_out;

  char* ws = (char*)d_ws;
  size_t o = 0;
  float*     ga    = (float*)(ws + o);     o = align256(o + sizeof(float) * (size_t)BB * NF * PIX);
  float*     cam   = (float*)(ws + o);     o = align256(o + sizeof(float) * (size_t)BB * NC * PIX);
  float*     wsc   = (float*)(ws + o);     o = align256(o + sizeof(float) * (size_t)BB * NC * WW);
  float*     hsc   = (float*)(ws + o);     o = align256(o + sizeof(float) * (size_t)BB * NC * HH);
  float*     gf    = (float*)(ws + o);     o = align256(o + sizeof(float) * (size_t)BB * NF);
  _Float16*  fw16p = (_Float16*)(ws + o);  o = align256(o + sizeof(_Float16) * (size_t)NF * KPAD);
  _Float16*  cw16  = (_Float16*)(ws + o);  o = align256(o + sizeof(_Float16) * (size_t)NC * NF);
  int*       boxes = (int*)(ws + o);       o = align256(o + sizeof(int) * (size_t)BB * TOPN * 4);
  float*     la    = (float*)(ws + o);     o = align256(o + sizeof(float) * (size_t)BB * TOPN * NF * PIX);
  float*     lf    = (float*)(ws + o);     o = align256(o + sizeof(float) * (size_t)BB * TOPN * NF);

  // weight conversion (done once; reused by every tile, stays in L2)
  k_cvt_pad<<<(NF * KPAD + 255) / 256, 256, 0, stream>>>(fw, fw16p, NF);
  k_cvt_f16<<<(NC * NF + 255) / 256, 256, 0, stream>>>(cw, cw16, NC * NF);

  // global branch
  k_conv_global<<<NGLOB / 16, 256, 0, stream>>>(in, fw16p, fb, ga);                    // 98 blocks
  k_reduce_mean<<<(BB * NF + 255) / 256, 256, 0, stream>>>(ga, gf, BB * NF);
  k_head_gs<<<(BB * NC + 63) / 64, 64, 0, stream>>>(gf, cw, cb, out);

  // CAM -> scores -> boxes
  k_cam<<<NGLOB / 16, 160, 0, stream>>>(ga, cw16, cb, cam);                            // 98 blocks
  k_scores<<<BB * NC, 448, 0, stream>>>(cam, wsc, hsc);                                // 640 blocks
  k_boxes<<<1, 32, 0, stream>>>(out, wsc, hsc, boxes);

  // local branch (crop sampling fused into conv im2col)
  k_conv_local<<<NLOC / 16, 256, 0, stream>>>(in, fw16p, fb, boxes, la);               // 392 blocks
  k_reduce_mean<<<(BB * TOPN * NF + 255) / 256, 256, 0, stream>>>(la, lf, BB * TOPN * NF);
  k_head_ls<<<(BB * NC + 63) / 64, 64, 0, stream>>>(lf, cw, cb, out);
}
